// ramsey_MPNN_13374528160302
// MI455X (gfx1250) — matrix operand order repaired and checked by reading, not compiled, not run on hardware
//
#include <hip/hip_runtime.h>
#include <hip/hip_bf16.h>

typedef __attribute__((ext_vector_type(16))) _Float16 v16h;
typedef __attribute__((ext_vector_type(8)))  float    v8f;

#define NN 2048
#define FF 32
#define HH 64
#define NEGS 0.01f

// ---------------- utility: zero via float4 ----------------
__global__ void k_zero(float* __restrict__ p, int n4) {
  float4 z = {0.f, 0.f, 0.f, 0.f};
  for (int i = blockIdx.x * blockDim.x + threadIdx.x; i < n4;
       i += gridDim.x * blockDim.x)
    ((float4*)p)[i] = z;
}

// ---------------- exclusive prefix-scan GIN input ----------------
// Y[n] = (1+eps)*X[n] + sum_{m<n} X[m], per channel. One block per channel.
__global__ void k_scan(const float* __restrict__ X, const float* __restrict__ epsP,
                       float* __restrict__ Y, int Cdim) {
  __shared__ float s[256];
  const int c = blockIdx.x;
  const int t = threadIdx.x;
  const int RPT = NN / 256;  // 8
  float eps1 = 1.0f + *epsP;
  float xv[NN / 256];
  float loc = 0.f;
  int row0 = t * RPT;
  #pragma unroll
  for (int r = 0; r < RPT; r++) { xv[r] = X[(row0 + r) * Cdim + c]; loc += xv[r]; }
  s[t] = loc;
  __syncthreads();
  for (int off = 1; off < 256; off <<= 1) {
    float v = (t >= off) ? s[t - off] : 0.f;
    __syncthreads();
    s[t] += v;
    __syncthreads();
  }
  float run = (t > 0) ? s[t - 1] : 0.f;  // exclusive prefix of thread blocks
  #pragma unroll
  for (int r = 0; r < RPT; r++) {
    Y[(row0 + r) * Cdim + c] = eps1 * xv[r] + run;
    run += xv[r];
  }
}

// ---------------- small dense MLP layer: Y = act(X@W + b) [+ resid] ----------
__global__ void k_mlp(const float* __restrict__ X, const float* __restrict__ W,
                      const float* __restrict__ b, const float* __restrict__ resid,
                      float* __restrict__ Y, int Cin, int Cout, int doRelu) {
  int id = blockIdx.x * blockDim.x + threadIdx.x;
  if (id >= NN * Cout) return;
  int n = id / Cout, co = id % Cout;
  const float* xr = X + n * Cin;
  float acc = b[co];
  for (int k = 0; k < Cin; k++) acc = fmaf(xr[k], W[k * Cout + co], acc);
  if (doRelu) acc = acc > 0.f ? acc : 0.f;
  if (resid) acc += resid[id];
  Y[id] = acc;
}

// ---------------- batchnorm over nodes: one block per channel -------------
__global__ void k_bnstats(const float* __restrict__ X, const float* __restrict__ g,
                          const float* __restrict__ bb, float* __restrict__ scale,
                          float* __restrict__ shift) {
  __shared__ float s1[256], s2[256];
  const int c = blockIdx.x, t = threadIdx.x;
  float a = 0.f, q = 0.f;
  for (int r = t; r < NN; r += 256) { float v = X[r * HH + c]; a += v; q += v * v; }
  s1[t] = a; s2[t] = q;
  __syncthreads();
  for (int off = 128; off > 0; off >>= 1) {
    if (t < off) { s1[t] += s1[t + off]; s2[t] += s2[t + off]; }
    __syncthreads();
  }
  if (t == 0) {
    float m = s1[0] / (float)NN;
    float v = s2[0] / (float)NN - m * m;
    float sc = g[c] * rsqrtf(v + 1e-5f);
    scale[c] = sc;
    shift[c] = bb[c] - m * sc;
  }
}

__global__ void k_bnapply(const float* __restrict__ X, const float* __restrict__ scale,
                          const float* __restrict__ shift, float* __restrict__ Y) {
  int id = blockIdx.x * blockDim.x + threadIdx.x;
  if (id < NN * HH) { int c = id % HH; Y[id] = X[id] * scale[c] + shift[c]; }
}

// ---------------- WMMA fragment builders (gfx1250 wave32 layouts) ----------
// A (f16 16x32, MxK): lane L -> row M=L&15; hi=L>>4; halfs t: K = (t<8?0:16)+hi*8+(t&7)
__device__ inline v16h build_a(const float* __restrict__ Hf,
                               const int* __restrict__ src, const int* __restrict__ dst,
                               int base, int lane, int E) {
  int r = lane & 15, hi = lane >> 4;
  int e = base + r; e = e < E ? e : E - 1;
  int i = src[e], j = dst[e];
  const float4* pi = (const float4*)(Hf + i * FF);
  const float4* pj = (const float4*)(Hf + j * FF);
  float4 a0 = pi[hi * 2 + 0], a1 = pi[hi * 2 + 1], a2 = pi[hi * 2 + 4], a3 = pi[hi * 2 + 5];
  float4 c0 = pj[hi * 2 + 0], c1 = pj[hi * 2 + 1], c2 = pj[hi * 2 + 4], c3 = pj[hi * 2 + 5];
  v16h a;
  a[0]  = (_Float16)(a0.x * c0.x); a[1]  = (_Float16)(a0.y * c0.y);
  a[2]  = (_Float16)(a0.z * c0.z); a[3]  = (_Float16)(a0.w * c0.w);
  a[4]  = (_Float16)(a1.x * c1.x); a[5]  = (_Float16)(a1.y * c1.y);
  a[6]  = (_Float16)(a1.z * c1.z); a[7]  = (_Float16)(a1.w * c1.w);
  a[8]  = (_Float16)(a2.x * c2.x); a[9]  = (_Float16)(a2.y * c2.y);
  a[10] = (_Float16)(a2.z * c2.z); a[11] = (_Float16)(a2.w * c2.w);
  a[12] = (_Float16)(a3.x * c3.x); a[13] = (_Float16)(a3.y * c3.y);
  a[14] = (_Float16)(a3.z * c3.z); a[15] = (_Float16)(a3.w * c3.w);
  return a;
}

// B (f16 32x16, KxN): lane L -> column N=(L&15)+c0; half t: K = (t<8?0:16)+hi*8+(t&7)
__device__ inline v16h build_b(const float* __restrict__ W5, int c0, int lane) {
  int r = lane & 15, hi = lane >> 4;
  v16h b;
  #pragma unroll
  for (int t = 0; t < 16; t++) b[t] = (_Float16)W5[((t < 8 ? 0 : 16) + hi * 8 + (t & 7)) * HH + c0 + r];
  return b;
}

// ---------------- edge pass A: WMMA + BN statistics ----------------
__global__ void __launch_bounds__(256)
k_edge_a(const float* __restrict__ Hf, const int* __restrict__ src,
         const int* __restrict__ dst, const float* __restrict__ W5,
         const float* __restrict__ b5, float* __restrict__ sumu,
         float* __restrict__ sumu2, int E) {
  __shared__ float ls1[HH], ls2[HH];
  const int t = threadIdx.x;
  if (t < HH) { ls1[t] = 0.f; ls2[t] = 0.f; }
  __syncthreads();
  const int lane = t & 31, r = lane & 15, hi = lane >> 4;
  const int wid = (blockIdx.x * blockDim.x + t) >> 5;
  const int wtot = (gridDim.x * blockDim.x) >> 5;
  const int numTiles = (E + 15) >> 4;
  v16h bf[4];
  float b5v[4];
  #pragma unroll
  for (int tt = 0; tt < 4; tt++) { bf[tt] = build_b(W5, tt * 16, lane); b5v[tt] = b5[tt * 16 + r]; }
  float su[4] = {0.f, 0.f, 0.f, 0.f}, sq[4] = {0.f, 0.f, 0.f, 0.f};
  for (int tile = wid; tile < numTiles; tile += wtot) {
    int base = tile * 16;
    // prefetch next tile's gather indices (serial dependence: idx -> h row -> WMMA)
    int nb = base + wtot * 16;
    if (nb < E) {
      __builtin_prefetch(&src[nb + r], 0, 1);
      __builtin_prefetch(&dst[nb + r], 0, 1);
    }
    v16h a = build_a(Hf, src, dst, base, lane, E);
    bool full = (base + 16) <= E;  // wave-uniform: only last tile can be partial
    #pragma unroll
    for (int tt = 0; tt < 4; tt++) {
      v8f cz = {};
      v8f d = __builtin_amdgcn_wmma_f32_16x16x32_f16(false, a, false, bf[tt],
                                                     (short)0, cz, false, false);
      if (full) {
        #pragma unroll
        for (int jj = 0; jj < 8; jj++) {
          float u = d[jj] + b5v[tt];
          u = u > 0.f ? u : NEGS * u;
          su[tt] += u; sq[tt] += u * u;
        }
      } else {
        #pragma unroll
        for (int jj = 0; jj < 8; jj++) {
          float u = d[jj] + b5v[tt];
          u = u > 0.f ? u : NEGS * u;
          if (base + hi * 8 + jj < E) { su[tt] += u; sq[tt] += u * u; }
        }
      }
    }
  }
  #pragma unroll
  for (int tt = 0; tt < 4; tt++) {
    atomicAdd(&ls1[tt * 16 + r], su[tt]);
    atomicAdd(&ls2[tt * 16 + r], sq[tt]);
  }
  __syncthreads();
  if (t < HH) { atomicAdd(&sumu[t], ls1[t]); atomicAdd(&sumu2[t], ls2[t]); }
}

// ---------------- fold edge-BN into W6/b6 ----------------
__global__ void k_fold(const float* __restrict__ sumu, const float* __restrict__ sumu2,
                       const float* __restrict__ g, const float* __restrict__ bb,
                       const float* __restrict__ W6, const float* __restrict__ b6,
                       float* __restrict__ W6f, float Einv) {
  __shared__ float sc[HH], sh[HH];
  const int t = threadIdx.x;  // 128 threads
  if (t < HH) {
    float m = sumu[t] * Einv;
    float v = sumu2[t] * Einv - m * m;
    float s = g[t] * rsqrtf(v + 1e-5f);
    sc[t] = s; sh[t] = bb[t] - m * s;
  }
  __syncthreads();
  if (t < 128) W6f[t] = sc[t >> 1] * W6[t];
  if (t < 2) {
    float acc = b6[t];
    for (int n = 0; n < HH; n++) acc += sh[n] * W6[2 * n + t];
    W6f[128 + t] = acc;
  }
}

// ---------------- edge pass B: WMMA + logits + softmax + scatter ----------
__global__ void __launch_bounds__(256)
k_edge_b(const float* __restrict__ Hf, const int* __restrict__ src,
         const int* __restrict__ dst, const float* __restrict__ W5,
         const float* __restrict__ b5, const float* __restrict__ W6f,
         float* __restrict__ out, int E) {
  __shared__ float lw6[130];
  __shared__ float lu[8][16][65];
  const int t = threadIdx.x;
  if (t < 130) lw6[t] = W6f[t];
  __syncthreads();
  const int lane = t & 31, wv = t >> 5, r = lane & 15, hi = lane >> 4;
  const int wid = (blockIdx.x * blockDim.x + t) >> 5;
  const int wtot = (gridDim.x * blockDim.x) >> 5;
  const int numTiles = (E + 15) >> 4;
  v16h bf[4];
  float b5v[4];
  #pragma unroll
  for (int tt = 0; tt < 4; tt++) { bf[tt] = build_b(W5, tt * 16, lane); b5v[tt] = b5[tt * 16 + r]; }
  for (int tile = wid; tile < numTiles; tile += wtot) {
    int base = tile * 16;
    int nb = base + wtot * 16;
    if (nb < E) {
      __builtin_prefetch(&src[nb + r], 0, 1);
      __builtin_prefetch(&dst[nb + r], 0, 1);
    }
    v16h a = build_a(Hf, src, dst, base, lane, E);
    #pragma unroll
    for (int tt = 0; tt < 4; tt++) {
      v8f cz = {};
      v8f d = __builtin_amdgcn_wmma_f32_16x16x32_f16(false, a, false, bf[tt],
                                                     (short)0, cz, false, false);
      #pragma unroll
      for (int jj = 0; jj < 8; jj++) {
        float u = d[jj] + b5v[tt];
        u = u > 0.f ? u : NEGS * u;
        lu[wv][hi * 8 + jj][tt * 16 + r] = u;  // stage u tile in LDS
      }
    }
    // DS ops from the same wave are in-order; fence keeps compiler honest.
    __builtin_amdgcn_fence(__ATOMIC_ACQ_REL, "wavefront");
    if (lane < 16) {
      int e = base + lane;
      if (e < E) {
        float l0 = lw6[128], l1 = lw6[129];
        #pragma unroll 8
        for (int n = 0; n < HH; n++) {
          float u = lu[wv][lane][n];
          l0 = fmaf(u, lw6[2 * n], l0);
          l1 = fmaf(u, lw6[2 * n + 1], l1);
        }
        float m = fmaxf(l0, l1);
        float e0 = __expf(l0 - m), e1 = __expf(l1 - m);
        float inv = 1.f / (e0 + e1);
        float2 pv = {e0 * inv, e1 * inv};
        int i = src[e], j = dst[e];
        ((float2*)out)[i * NN + j] = pv;
        ((float2*)out)[j * NN + i] = pv;
      }
    }
  }
}

extern "C" void kernel_launch(void* const* d_in, const int* in_sizes, int n_in,
                              void* d_out, int out_size, void* d_ws, size_t ws_size,
                              hipStream_t stream) {
  const float* x_nf  = (const float*)d_in[1];
  const float* eps1  = (const float*)d_in[2];
  const float* W1a   = (const float*)d_in[3];
  const float* b1a   = (const float*)d_in[4];
  const float* W1b   = (const float*)d_in[5];
  const float* b1b   = (const float*)d_in[6];
  const float* epsm  = (const float*)d_in[7];
  const float* Wma   = (const float*)d_in[8];
  const float* bma   = (const float*)d_in[9];
  const float* Wmb   = (const float*)d_in[10];
  const float* bmb   = (const float*)d_in[11];
  const float* bnm_g = (const float*)d_in[12];
  const float* bnm_b = (const float*)d_in[13];
  const float* epsl  = (const float*)d_in[14];
  const float* Wla   = (const float*)d_in[15];
  const float* bla   = (const float*)d_in[16];
  const float* Wlb   = (const float*)d_in[17];
  const float* blb   = (const float*)d_in[18];
  const float* W5    = (const float*)d_in[19];
  const float* b5    = (const float*)d_in[20];
  const float* bn5_g = (const float*)d_in[21];
  const float* bn5_b = (const float*)d_in[22];
  const float* W6    = (const float*)d_in[23];
  const float* b6    = (const float*)d_in[24];
  const int*   esrc  = (const int*)d_in[25];
  const int*   edst  = (const int*)d_in[26];
  const int E = in_sizes[25];
  float* out = (float*)d_out;

  float* ws    = (float*)d_ws;
  float* buf0  = ws;                 // N*64
  float* buf1  = buf0 + NN * HH;     // N*64
  float* buf2  = buf1 + NN * HH;     // N*64
  float* buf3  = buf2 + NN * HH;     // N*32 (final h)
  float* small = buf3 + NN * FF;
  float* bnsc  = small;              // 64
  float* bnshf = small + 64;         // 64
  float* sumu  = small + 128;        // 64
  float* sumu2 = small + 192;        // 64
  float* W6f   = small + 256;        // 130

  // zero output (diagonal must stay 0) and edge-BN accumulators
  k_zero<<<2048, 256, 0, stream>>>(out, (NN * NN * 2) / 4);
  k_zero<<<1, 32, 0, stream>>>(sumu, 32);

  // conv1: scan -> mlp -> mlp (leaky after relu is identity)
  k_scan<<<FF, 256, 0, stream>>>(x_nf, eps1, buf0, FF);
  k_mlp<<<(NN * HH) / 256, 256, 0, stream>>>(buf0, W1a, b1a, nullptr, buf1, FF, HH, 1);
  k_mlp<<<(NN * HH) / 256, 256, 0, stream>>>(buf1, W1b, b1b, nullptr, buf2, HH, HH, 1);
  // middle conv + BN
  k_scan<<<HH, 256, 0, stream>>>(buf2, epsm, buf0, HH);
  k_mlp<<<(NN * HH) / 256, 256, 0, stream>>>(buf0, Wma, bma, nullptr, buf1, HH, HH, 1);
  k_mlp<<<(NN * HH) / 256, 256, 0, stream>>>(buf1, Wmb, bmb, nullptr, buf0, HH, HH, 1);
  k_bnstats<<<HH, 256, 0, stream>>>(buf0, bnm_g, bnm_b, bnsc, bnshf);
  k_bnapply<<<(NN * HH) / 256, 256, 0, stream>>>(buf0, bnsc, bnshf, buf2);
  // last conv + residual
  k_scan<<<HH, 256, 0, stream>>>(buf2, epsl, buf0, HH);
  k_mlp<<<(NN * HH) / 256, 256, 0, stream>>>(buf0, Wla, bla, nullptr, buf1, HH, HH, 1);
  k_mlp<<<(NN * FF) / 256, 256, 0, stream>>>(buf1, Wlb, blb, x_nf, buf3, HH, FF, 1);
  // edge predictor: WMMA pass A (stats) -> fold BN -> WMMA pass B (output)
  k_edge_a<<<1024, 256, 0, stream>>>(buf3, esrc, edst, W5, b5, sumu, sumu2, E);
  k_fold<<<1, 128, 0, stream>>>(sumu, sumu2, bn5_g, bn5_b, W6, b6, W6f, 1.0f / (float)E);
  k_edge_b<<<1024, 256, 0, stream>>>(buf3, esrc, edst, W5, b5, W6f, out, E);
}
